// RAttention_Diff_52682068853336
// MI455X (gfx1250) — matrix operand order repaired and checked by reading, not compiled, not run on hardware
//
#include <hip/hip_runtime.h>
#include <hip/hip_bf16.h>
#include <cstdint>

// ---------------------------------------------------------------------------
// RAttention (criss-cross attention) for MI455X / gfx1250, wave32 + WMMA.
//
// All heavy einsums are mapped onto v_wmma_f32_16x16x32_bf16 (f32 accum).
// The LR/RL branches of the reference are softmax over a singleton axis
// (identically 1.0) -> replaced by a constant +2.0 in the finalize step.
// The dominant conv1x1 GEMM stages its activation tiles through LDS with the
// Tensor Data Mover (tensor_load_to_lds, TENSORcnt), double-buffered.
// ---------------------------------------------------------------------------

typedef __bf16 bf16_t;
typedef __attribute__((ext_vector_type(16))) __bf16 v16bf;
typedef __attribute__((ext_vector_type(8)))  float  v8f;
typedef __attribute__((ext_vector_type(4)))  unsigned int u32x4;
typedef __attribute__((ext_vector_type(8)))  int i32x8;
typedef __attribute__((ext_vector_type(4)))  int i32x4;

#define NB 8
#define CC 512
#define C8 64
#define HH 96
#define WW 96
#define HW 9216      // H*W
#define LL 192       // H+W

#if defined(__HIP_DEVICE_COMPILE__) && __has_builtin(__builtin_amdgcn_tensor_load_to_lds)
#define HAVE_TDM 1
#else
#define HAVE_TDM 0
#endif

static __device__ __forceinline__ v8f wmma_bf16(v16bf a, v16bf b, v8f c) {
  // (neg_a, A, neg_b, B, c_mod, C, reuse_a, reuse_b)
  return __builtin_amdgcn_wmma_f32_16x16x32_bf16(false, a, false, b, (short)0, c,
                                                 false, false);
}

// K offset of the low element held in A-fragment VGPR v (contiguous order, matches B):
// lanes 0-15 : V0..V7 -> K 0..15
// lanes 16-31: V0..V7 -> K 16..31   (add hi*16)
static __device__ __forceinline__ int a_kbase(int v, int hi) {
  return 2 * v + hi * 16;
}

#if HAVE_TDM
// Issue a TDM load of a 2D f32 tile (tile_w x tile_h, row stride in elements)
// from global memory into LDS at byte offset lds_off. One issue per *wave*;
// caller must ensure only one wave executes this (TDM ignores EXEC).
static __device__ __forceinline__ void tdm_load_2d_f32(
    const float* gptr, unsigned lds_off, unsigned tile_w, unsigned tile_h,
    unsigned row_stride_elems)
{
  const unsigned long long ga = (unsigned long long)(uintptr_t)gptr;
  u32x4 g0;
  g0[0] = 1u;                                            // count=1 (valid D#)
  g0[1] = lds_off;                                       // lds_addr (bytes)
  g0[2] = (unsigned)(ga & 0xffffffffu);                  // global_addr[31:0]
  g0[3] = (unsigned)((ga >> 32) & 0x1ffffffu)            // global_addr[56:32]
        | (2u << 30);                                    // type = 2 ("image")
  i32x8 g1;
  g1[0] = (int)(2u << 16);                               // data_size=2 (4 bytes)
  g1[1] = (int)((row_stride_elems & 0xffffu) << 16);     // tensor_dim0 lo16
  g1[2] = (int)((row_stride_elems >> 16) & 0xffffu)      // tensor_dim0 hi16
        | (int)((CC & 0xffffu) << 16);                   // tensor_dim1 lo16
  g1[3] = (int)((tile_w & 0xffffu) << 16);               // dim1 hi16=0, tile_dim0
  g1[4] = (int)(tile_h & 0xffffu);                       // tile_dim1; tile_dim2=0
  g1[5] = (int)row_stride_elems;                         // tensor_dim0_stride[31:0]
  g1[6] = 0;                                             // stride hi / dim1_stride
  g1[7] = 0;
  const i32x4 z4 = {0, 0, 0, 0};                         // groups 2/3 unused (2D)
  const i32x8 z8 = {0, 0, 0, 0, 0, 0, 0, 0};
  __builtin_amdgcn_tensor_load_to_lds(g0, g1, z4, z4, z8, 0);
}
#endif

// One K=32 WMMA step of the conv GEMM: A from global weights, B from LDS tile.
static __device__ __forceinline__ v8f conv_step(const float* __restrict__ wrow,
                                                const float* __restrict__ xs,
                                                int hi, v8f acc)
{
  v16bf a, bf;
#pragma unroll
  for (int v = 0; v < 8; ++v) {
    int kk = a_kbase(v, hi);
    a[2 * v + 0] = (bf16_t)wrow[kk + 0];
    a[2 * v + 1] = (bf16_t)wrow[kk + 1];
  }
#pragma unroll
  for (int v = 0; v < 8; ++v) {
    int k = hi * 16 + 2 * v;
    bf[2 * v + 0] = (bf16_t)xs[(k + 0) * 128];
    bf[2 * v + 1] = (bf16_t)xs[(k + 1) * 128];
  }
  return wmma_bf16(a, bf, acc);
}

// ---------------------------------------------------------------------------
// 1x1 conv as GEMM: O[b,m,p] = sum_k Wt[m,k] * X[b,k,p] + bias[m], bf16 out.
// Block = 8 waves; each wave owns a 16(M)x16(P) tile; block tile = 16x128.
// X tiles (32x128 f32) staged in LDS by the Tensor Data Mover, double-buffered.
// ---------------------------------------------------------------------------
__global__ __launch_bounds__(256) void conv1x1_gemm(
    const float* __restrict__ X,    // (B, 512, 9216)
    const float* __restrict__ Wt,   // (M, 512)
    const float* __restrict__ bias, // (M)
    bf16_t* __restrict__ O,         // (B, M, 9216)
    int M)
{
  __shared__ float xtile[2][32 * 128];   // 2 x 16 KB

  const int lane = threadIdx.x, hl = lane & 15, hi = lane >> 4;
  const int b  = blockIdx.z;
  const int m0 = blockIdx.y * 16;
  const int p0 = blockIdx.x * 128;

  const float* Xb = X + (size_t)b * CC * HW + p0;   // (k=0, p0) of this block
  v8f acc = {};

#if HAVE_TDM
  const unsigned lds0 = (unsigned)(uintptr_t)&xtile[0][0];
  const unsigned lds1 = (unsigned)(uintptr_t)&xtile[1][0];
  const int wid = __builtin_amdgcn_readfirstlane(threadIdx.y);
  if (wid == 0)                                   // scalar branch: 1 issue/block
    tdm_load_2d_f32(Xb, lds0, 128, 32, HW);
  int cur = 0;
  for (int k0 = 0; k0 < CC; k0 += 32) {
    if (wid == 0) {
      if (k0 + 32 < CC) {                         // prefetch next tile
        tdm_load_2d_f32(Xb + (size_t)(k0 + 32) * HW, cur ? lds0 : lds1, 128, 32, HW);
        __builtin_amdgcn_s_wait_tensorcnt(1);     // current tile landed
      } else {
        __builtin_amdgcn_s_wait_tensorcnt(0);     // last tile landed
      }
    }
    __syncthreads();                              // LDS tile visible to all waves
    acc = conv_step(Wt + (size_t)(m0 + hl) * CC + k0,
                    &xtile[cur][0] + threadIdx.y * 16 + hl, hi, acc);
    __syncthreads();                              // done reading before overwrite
    cur ^= 1;
  }
#else
  const int tid = threadIdx.y * 32 + threadIdx.x; // 0..255
  for (int k0 = 0; k0 < CC; k0 += 32) {
#pragma unroll
    for (int t = 0; t < 16; ++t) {                // cooperative 32x128 tile load
      int e = tid + 256 * t;
      xtile[0][e] = Xb[(size_t)(k0 + (e >> 7)) * HW + (e & 127)];
    }
    __syncthreads();
    acc = conv_step(Wt + (size_t)(m0 + hl) * CC + k0,
                    &xtile[0][0] + threadIdx.y * 16 + hl, hi, acc);
    __syncthreads();
  }
#endif

#pragma unroll
  for (int r = 0; r < 8; ++r) {
    int m = m0 + r + hi * 8;
    O[((size_t)b * M + m) * HW + (p0 + threadIdx.y * 16 + hl)] =
        (bf16_t)(acc[r] + bias[m]);
  }
}

// ---------------------------------------------------------------------------
// Attention logits. mode 0: eH(h,j) = sum_c q[b,c,h,s]*k[b,c,j,s]   (s = w),
//                   with h==j masked to -inf, written to logits[...,j].
//                   mode 1: eW(w,j) = sum_c q[b,c,s,w]*k[b,c,s,j]   (s = h),
//                   written to logits[...,96+j].
// ---------------------------------------------------------------------------
__global__ __launch_bounds__(192) void attn_logits(
    const bf16_t* __restrict__ Q,   // (B, 64, 96, 96)
    const bf16_t* __restrict__ Km,  // (B, 64, 96, 96)
    float* __restrict__ logits,     // (B, 96, 96, 192)
    int mode)
{
  const int lane = threadIdx.x, hl = lane & 15, hi = lane >> 4;
  const int m0 = threadIdx.y * 16;
  const int n0 = blockIdx.x * 16;
  const int s  = blockIdx.y;
  const int b  = blockIdx.z;
  const size_t base = (size_t)b * C8 * HW;

  v8f acc = {};
  for (int c0 = 0; c0 < C8; c0 += 32) {
    v16bf a, bf;
#pragma unroll
    for (int v = 0; v < 8; ++v) {       // A row = m0+hl, K = channel (stride HW)
      int ch = c0 + a_kbase(v, hi);
      size_t i0 = (mode == 0)
          ? base + (size_t)ch * HW + (size_t)(m0 + hl) * WW + s
          : base + (size_t)ch * HW + (size_t)s * WW + (m0 + hl);
      a[2 * v + 0] = Q[i0];
      a[2 * v + 1] = Q[i0 + HW];
    }
#pragma unroll
    for (int v = 0; v < 8; ++v) {       // B col = n0+hl, K = channel
      int ch = c0 + hi * 16 + 2 * v;
      size_t i0 = (mode == 0)
          ? base + (size_t)ch * HW + (size_t)(n0 + hl) * WW + s
          : base + (size_t)ch * HW + (size_t)s * WW + (n0 + hl);
      bf[2 * v + 0] = Km[i0];
      bf[2 * v + 1] = Km[i0 + HW];
    }
    acc = wmma_bf16(a, bf, acc);
  }
#pragma unroll
  for (int r = 0; r < 8; ++r) {
    int mrow = m0 + r + hi * 8;
    int j = n0 + hl;
    float val = acc[r];
    size_t o;
    if (mode == 0) {
      if (mrow == j) val = -__builtin_inff();
      o = (((size_t)b * HH + mrow) * WW + s) * LL + j;
    } else {
      o = (((size_t)b * HH + s) * WW + mrow) * LL + 96 + j;
    }
    logits[o] = val;
  }
}

// ---------------------------------------------------------------------------
// Softmax over L=192 per (b,h,w) row; one wave per row, 6 elems per lane.
// ---------------------------------------------------------------------------
__global__ __launch_bounds__(256) void softmax192(
    const float* __restrict__ logits, bf16_t* __restrict__ att)
{
  const int lane = threadIdx.x & 31;
  const size_t row = (size_t)blockIdx.x * 8 + (threadIdx.x >> 5);
  const float* in = logits + row * LL;

  float v[6];
  float mx = -__builtin_inff();
#pragma unroll
  for (int i = 0; i < 6; ++i) { v[i] = in[lane + 32 * i]; mx = fmaxf(mx, v[i]); }
#pragma unroll
  for (int o = 16; o; o >>= 1) mx = fmaxf(mx, __shfl_xor(mx, o, 32));
  float s = 0.f;
#pragma unroll
  for (int i = 0; i < 6; ++i) { v[i] = __expf(v[i] - mx); s += v[i]; }
#pragma unroll
  for (int o = 16; o; o >>= 1) s += __shfl_xor(s, o, 32);
  const float inv = 1.0f / s;
  bf16_t* out = att + row * LL;
#pragma unroll
  for (int i = 0; i < 6; ++i) out[lane + 32 * i] = (bf16_t)(v[i] * inv);
}

// ---------------------------------------------------------------------------
// Attention apply. mode 0: Out[b,c,n,s] (+)= sum_j V[b,c,j,s]  * att[b,n,s,j]
//                  mode 1: Out[b,c,s,n] (+)= sum_j V[b,c,s,j]  * att[b,s,n,96+j]
// Block = 8 waves, each a 16(C)x16(N) tile; blockIdx.z = b*4 + cgroup.
// ---------------------------------------------------------------------------
__global__ __launch_bounds__(256) void apply_att(
    const bf16_t* __restrict__ V,    // (B, 512, 96, 96)
    const bf16_t* __restrict__ att,  // (B, 96, 96, 192)
    float* __restrict__ acc,         // (B, 512, 96, 96)
    int mode, int accumulate)
{
  const int lane = threadIdx.x, hl = lane & 15, hi = lane >> 4;
  const int b  = blockIdx.z >> 2;
  const int c0 = ((blockIdx.z & 3) * 8 + threadIdx.y) * 16;
  const int n0 = blockIdx.x * 16;    // h (mode0) / w (mode1)
  const int s  = blockIdx.y;         // w (mode0) / h (mode1)

  const size_t vbase = (size_t)b * CC * HW;
  const size_t abase = (size_t)b * HH * WW * LL;

  v8f c = {};
  for (int j0 = 0; j0 < 96; j0 += 32) {
    v16bf a, bf;
#pragma unroll
    for (int v = 0; v < 8; ++v) {     // A row = channel c0+hl, K = j
      int j = j0 + a_kbase(v, hi);
      size_t i0 = (mode == 0)
          ? vbase + (size_t)(c0 + hl) * HW + (size_t)j * WW + s
          : vbase + (size_t)(c0 + hl) * HW + (size_t)s * WW + j;
      a[2 * v + 0] = V[i0];
      a[2 * v + 1] = V[i0 + (mode == 0 ? WW : 1)];
    }
#pragma unroll
    for (int v = 0; v < 8; ++v) {     // B col = n0+hl, K = j (contiguous)
      int j = j0 + hi * 16 + 2 * v;
      size_t i0 = (mode == 0)
          ? abase + ((size_t)(n0 + hl) * WW + s) * LL + j
          : abase + ((size_t)s * WW + (n0 + hl)) * LL + 96 + j;
      bf[2 * v + 0] = att[i0];
      bf[2 * v + 1] = att[i0 + 1];
    }
    c = wmma_bf16(a, bf, c);
  }
#pragma unroll
  for (int r = 0; r < 8; ++r) {
    int ch = c0 + r + hi * 8;
    size_t o = (mode == 0)
        ? vbase + (size_t)ch * HW + (size_t)(n0 + hl) * WW + s
        : vbase + (size_t)ch * HW + (size_t)s * WW + (n0 + hl);
    float val = c[r];
    if (accumulate) val += acc[o];
    acc[o] = val;
  }
}

// ---------------------------------------------------------------------------
// y = gamma * (acc + 2.0) + x   (the +2 absorbs the degenerate LR/RL paths)
// ---------------------------------------------------------------------------
__global__ __launch_bounds__(256) void finalize_out(
    const float* __restrict__ acc, const float* __restrict__ x,
    const float* __restrict__ gamma, float* __restrict__ y, long long n4)
{
  const float g = gamma[0];
  const float4* a4 = (const float4*)acc;
  const float4* x4 = (const float4*)x;
  float4* y4 = (float4*)y;
  for (long long i = (long long)blockIdx.x * blockDim.x + threadIdx.x; i < n4;
       i += (long long)gridDim.x * blockDim.x) {
    float4 a = a4[i], xx = x4[i], r;
    r.x = g * (a.x + 2.0f) + xx.x;
    r.y = g * (a.y + 2.0f) + xx.y;
    r.z = g * (a.z + 2.0f) + xx.z;
    r.w = g * (a.w + 2.0f) + xx.w;
    y4[i] = r;
  }
}

// ---------------------------------------------------------------------------
extern "C" void kernel_launch(void* const* d_in, const int* in_sizes, int n_in,
                              void* d_out, int out_size, void* d_ws, size_t ws_size,
                              hipStream_t stream) {
  const float* x_ex = (const float*)d_in[0];
  const float* x_q  = (const float*)d_in[1];
  const float* Wq   = (const float*)d_in[2];
  const float* bq   = (const float*)d_in[3];
  const float* Wk   = (const float*)d_in[4];
  const float* bk   = (const float*)d_in[5];
  const float* Wv   = (const float*)d_in[6];
  const float* bv   = (const float*)d_in[7];
  const float* Wqv  = (const float*)d_in[8];
  const float* bqv  = (const float*)d_in[9];
  const float* g1   = (const float*)d_in[10];
  const float* g2   = (const float*)d_in[11];

  const size_t nOut = (size_t)NB * CC * HW;          // 37,748,736
  float* out_ex = (float*)d_out;
  float* out_q  = out_ex + nOut;

  // Workspace carve-up (bf16 intermediates; logits region reused as f32 acc)
  char* ws = (char*)d_ws;
  size_t off = 0;
  bf16_t* qb  = (bf16_t*)(ws + off); off += (size_t)NB * C8 * HW * sizeof(bf16_t);
  bf16_t* kb  = (bf16_t*)(ws + off); off += (size_t)NB * C8 * HW * sizeof(bf16_t);
  bf16_t* vb  = (bf16_t*)(ws + off); off += (size_t)NB * CC * HW * sizeof(bf16_t);
  bf16_t* qvb = (bf16_t*)(ws + off); off += (size_t)NB * CC * HW * sizeof(bf16_t);
  bf16_t* att = (bf16_t*)(ws + off); off += (size_t)NB * HH * WW * LL * sizeof(bf16_t);
  float*  logits = (float*)(ws + off);   // (B,96,96,192) f32, then reused:
  float*  acc    = logits;               // (B,512,96,96) f32 accumulator

  const dim3 convBlk(32, 8);
  // 1x1 convs -> bf16 q, k, v, qv
  conv1x1_gemm<<<dim3(72, C8 / 16, NB), convBlk, 0, stream>>>(x_q,  Wq,  bq,  qb,  C8);
  conv1x1_gemm<<<dim3(72, C8 / 16, NB), convBlk, 0, stream>>>(x_ex, Wk,  bk,  kb,  C8);
  conv1x1_gemm<<<dim3(72, CC / 16, NB), convBlk, 0, stream>>>(x_ex, Wv,  bv,  vb,  CC);
  conv1x1_gemm<<<dim3(72, CC / 16, NB), convBlk, 0, stream>>>(x_q,  Wqv, bqv, qvb, CC);

  // logits (eH with diag mask, eW), then joint softmax over 192
  attn_logits<<<dim3(6, WW, NB), dim3(32, 6), 0, stream>>>(qb, kb, logits, 0);
  attn_logits<<<dim3(6, HH, NB), dim3(32, 6), 0, stream>>>(qb, kb, logits, 1);
  softmax192<<<dim3((NB * HH * WW) / 8), dim3(256), 0, stream>>>(logits, att);

  // examplar path: out_H (write) + out_W (accumulate) -> finalize
  apply_att<<<dim3(6, WW, NB * 4), convBlk, 0, stream>>>(vb, att, acc, 0, 0);
  apply_att<<<dim3(6, HH, NB * 4), convBlk, 0, stream>>>(vb, att, acc, 1, 1);
  finalize_out<<<dim3(2304), dim3(256), 0, stream>>>(acc, x_ex, g1, out_ex,
                                                     (long long)(nOut / 4));

  // query path reuses the accumulator
  apply_att<<<dim3(6, WW, NB * 4), convBlk, 0, stream>>>(qvb, att, acc, 0, 0);
  apply_att<<<dim3(6, HH, NB * 4), convBlk, 0, stream>>>(qvb, att, acc, 1, 1);
  finalize_out<<<dim3(2304), dim3(256), 0, stream>>>(acc, x_q, g2, out_q,
                                                     (long long)(nOut / 4));
}
